// Semantic_Mapping_59304908423453
// MI455X (gfx1250) — matrix operand order repaired and checked by reading, not compiled, not run on hardware
//
#include <hip/hip_runtime.h>
#include <hip/hip_bf16.h>
#include <math.h>

typedef __attribute__((ext_vector_type(16))) _Float16 v16h;
typedef __attribute__((ext_vector_type(8)))  float    v8f;

namespace {
constexpr int   kB   = 8;
constexpr int   kH   = 120, kW = 160;
constexpr int   kC   = 20;            // obs / map channels
constexpr int   kF   = 17;            // splat feature channels (1 + 16 sem)
constexpr int   kVR  = 100;           // vision range cells
constexpr int   kZB  = 80;            // z bins
constexpr int   kMD  = 240;           // map dim
constexpr int   kP   = kH * kW;       // 19200 points
constexpr float kFL  = 97.0478f;      // W/2 / tan(39.5 deg)

// ---- workspace layout (floats) ----
constexpr size_t VOX_N   = (size_t)kF * kVR * kVR * kZB;   // 13,600,000
constexpr size_t WS_VOX  = 0;
constexpr size_t PROJ_N  = (size_t)kC * kVR * kVR;         // 200,000
constexpr size_t WS_PROJ = WS_VOX + VOX_N;
constexpr size_t ROT_N   = (size_t)kC * kMD * kMD;         // 1,152,000
constexpr size_t WS_ROT  = WS_PROJ + PROJ_N;
constexpr size_t WS_PAR  = WS_ROT + ROT_N;                 // 8 * 4 params

// ---- output layout (floats) ----
constexpr size_t OUT_FP  = 0;                              // (8,1,100,100)
constexpr size_t OUT_MAP = (size_t)kB * kVR * kVR;         // (8,20,240,240)
constexpr size_t OUT_P1  = OUT_MAP + (size_t)kB * kC * kMD * kMD;
constexpr size_t OUT_P2  = OUT_P1 + (size_t)kB * 3;
} // namespace

// ---------------------------------------------------------------- zero scratch
__global__ void k_zero(float4* __restrict__ p, int n4) {
  int t = blockIdx.x * blockDim.x + threadIdx.x;
  if (t < n4) p[t] = make_float4(0.f, 0.f, 0.f, 0.f);
}

// ---------------------------------------------------------------- pose update
__global__ void k_pose(const float* __restrict__ pose_obs,
                       const float* __restrict__ poses_last,
                       float* __restrict__ out, float* __restrict__ params) {
  int b = threadIdx.x;
  if (b >= kB) return;
  const float R2D = 57.29577951308232f;
  const float D2R = (float)(1.0 / 57.29577951308232);
  float px = poses_last[b * 3 + 0], py = poses_last[b * 3 + 1], pt = poses_last[b * 3 + 2];
  float ox = pose_obs[b * 3 + 0],   oy = pose_obs[b * 3 + 1],   ot = pose_obs[b * 3 + 2];
  float th = pt * D2R;
  float s = sinf(th), c = cosf(th);
  float ny = py + ox * s + oy * c;
  float nx = px + ox * c - oy * s;
  float nt = pt + ot * R2D;
  nt = fmodf(nt - 180.0f, 360.0f) + 180.0f;
  nt = fmodf(nt + 180.0f, 360.0f) - 180.0f;
  out[OUT_P1 + b * 3 + 0] = nx; out[OUT_P1 + b * 3 + 1] = ny; out[OUT_P1 + b * 3 + 2] = nt;
  out[OUT_P2 + b * 3 + 0] = nx; out[OUT_P2 + b * 3 + 1] = ny; out[OUT_P2 + b * 3 + 2] = nt;
  // transform params
  float st_t = 90.0f - nt;
  float ang  = st_t * (float)(3.14159265358979323846 / 180.0);
  float tx = -(nx * 20.0f - 120.0f) / 120.0f;
  float ty = -(ny * 20.0f - 120.0f) / 120.0f;
  params[b * 4 + 0] = cosf(ang);
  params[b * 4 + 1] = sinf(ang);
  params[b * 4 + 2] = tx * 119.5f;   // pixel shift x
  params[b * 4 + 3] = ty * 119.5f;   // pixel shift y
}

// ---------------------------------------------------------------- trilinear splat
__global__ void k_splat(const float* __restrict__ obs, float* __restrict__ vox, int b) {
  int t = blockIdx.x * blockDim.x + threadIdx.x;
  if (t >= kP * kF) return;
  int p = t % kP;          // coalesced point index within a wave
  int f = t / kP;
  int h = p / kW, w = p % kW;

  float depth = obs[(((size_t)b * kC + 3) * kH + h) * kW + w];
  float X   = (w - 79.5f) * depth / kFL;
  float pcz = (59.5f - h) * depth / kFL + 88.0f;
  float pcx = X + 250.0f;
  // follow the reference normalization chain, then de-normalize
  float xs = (pcx / 5.0f - 50.0f) / 100.0f * 2.0f;
  float ys = (depth / 5.0f - 50.0f) / 100.0f * 2.0f;
  float zs = (pcz / 5.0f - 32.0f) / 80.0f * 2.0f;
  float pos[3] = { xs * 50.0f + 50.0f, ys * 50.0f + 50.0f, zs * 40.0f + 40.0f };
  const int gd[3] = { kVR, kVR, kZB };

  int   idc[3][2];
  float wtc[3][2];
#pragma unroll
  for (int d = 0; d < 3; ++d) {
    float fl = floorf(pos[d]);
#pragma unroll
    for (int ix = 0; ix < 2; ++ix) {
      float pi   = fl + (float)ix;
      bool  safe = (pi > 0.0f) && (pi < (float)gd[d]);
      float wv   = (1.0f - fabsf(pos[d] - pi)) * (safe ? 1.0f : 0.0f);
      idc[d][ix] = safe ? (int)pi : 0;
      wtc[d][ix] = wv;
    }
  }
  float featv = (f == 0) ? 1.0f
                         : obs[(((size_t)b * kC + 3 + f) * kH + h) * kW + w];
  float* vf = vox + (size_t)f * kVR * kVR * kZB;
#pragma unroll
  for (int cx = 0; cx < 2; ++cx)
#pragma unroll
    for (int cy = 0; cy < 2; ++cy)
#pragma unroll
      for (int cz = 0; cz < 2; ++cz) {
        float wv = wtc[0][cx] * wtc[1][cy] * wtc[2][cz];
        if (wv > 0.0f) {
          int idx = (idc[0][cx] * kVR + idc[1][cy]) * kZB + idc[2][cz];
          atomicAdd(&vf[idx], featv * wv);
        }
      }
}

// ---------------------------------------------------------------- WMMA z-projection
// One wave per (feature f, 16-cell tile). A = round(vox) tile (16 cells x 32 z, f16),
// B col0 = full-range mask, col1 = window mask [9,35). D col0/1 = the two sums.
__global__ void k_proj(const float* __restrict__ vox, float* __restrict__ proj,
                       float* __restrict__ out, int b) {
  int wid  = blockIdx.x;              // 0 .. 17*625-1
  int f    = wid / 625;
  int tile = wid % 625;
  int lane = threadIdx.x;             // 0..31
  int row  = lane & 15;
  int cell = tile * 16 + row;         // linearized (x*100 + y)
  const float* base = vox + (size_t)f * kVR * kVR * kZB + (size_t)cell * kZB;
  int kofsA = (lane < 16) ? 0 : 8;    // A layout: low lanes K 0..7/16..23, high 8..15/24..31
  int kofsB = (lane < 16) ? 0 : 8;    // B layout: low lanes K 0..7/16..23, high 8..15/24..31
  int N     = lane & 15;              // B/D column

  // branch-free per-lane column flags (avoid EXEC-mask branches around WMMA)
  float isC0 = (N == 0) ? 1.0f : 0.0f;
  float isC1 = (N == 1) ? 1.0f : 0.0f;

  v8f acc = {};
  for (int chunk = 0; chunk < 3; ++chunk) {
    int kc = chunk * 32;
    v16h a;
#pragma unroll
    for (int j = 0; j < 8; ++j) {
      int z1 = kc + kofsA + j;
      int z2 = z1 + 16;
      float v1 = (z1 < kZB) ? rintf(base[z1]) : 0.0f;   // round-half-even per ref
      float v2 = (z2 < kZB) ? rintf(base[z2]) : 0.0f;
      a[j]     = (_Float16)v1;
      a[8 + j] = (_Float16)v2;
    }
    v16h bm;
#pragma unroll
    for (int j = 0; j < 16; ++j) {
      int z = kc + kofsB + (j < 8 ? j : j + 8);
      float full = (z < kZB) ? 1.0f : 0.0f;
      float win  = (z >= 9 && z < 35) ? 1.0f : 0.0f;
      bm[j] = (_Float16)(isC0 * full + isC1 * win);     // pure selects/FMA, no branches
    }
    acc = __builtin_amdgcn_wmma_f32_16x16x32_f16(false, a, false, bm,
                                                 (short)0, acc, false, false);
  }

  // D layout: VGPR r holds (M=r, N=lane) for lanes 0-15, (M=r+8, N=lane-16) for 16-31.
  int mofs = (lane >= 16) ? 8 : 0;
  if (N == 0) {                       // full-range sums -> explored map (f==0 only)
    if (f == 0) {
#pragma unroll
      for (int r = 0; r < 8; ++r) {
        int cc  = tile * 16 + r + mofs;       // = x*100 + y
        int pos = (cc % 100) * 100 + (cc / 100); // proj stored [y][x]
        proj[1 * 10000 + pos] = fminf(fmaxf(acc[r], 0.0f), 1.0f);
      }
    }
  } else if (N == 1) {                // window sums
#pragma unroll
    for (int r = 0; r < 8; ++r) {
      int cc  = tile * 16 + r + mofs;
      int pos = (cc % 100) * 100 + (cc / 100);
      float wv = acc[r];
      if (f == 0) {
        float v = fminf(fmaxf(wv, 0.0f), 1.0f);
        proj[pos] = v;                                   // agent_view ch 0
        out[OUT_FP + (size_t)b * 10000 + pos] = v;       // fp_map_pred output
      } else {
        proj[(size_t)(3 + f) * 10000 + pos] =
            fminf(fmaxf(wv * 0.2f, 0.0f), 1.0f);         // agent_view ch 4..19
      }
    }
  }
}

// ---------------------------------------------------------------- rotation resample
// agent_view is analytic: zero everywhere except [120:220,70:170] from proj planes.
__global__ void k_rot(const float* __restrict__ proj, const float* __restrict__ params,
                      float* __restrict__ rotated, int b) {
  int t = blockIdx.x * blockDim.x + threadIdx.x;
  if (t >= kMD * kMD) return;
  int i = t / kMD, j = t % kMD;
  float ct = params[b * 4 + 0], st = params[b * 4 + 1];
  float xn = (2.0f / 239.0f) * (float)j - 1.0f;
  float yn = (2.0f / 239.0f) * (float)i - 1.0f;
  float gx = ct * xn - st * yn;
  float gy = st * xn + ct * yn;
  float xs = (gx + 1.0f) * 119.5f;
  float ys = (gy + 1.0f) * 119.5f;
  float x0f = floorf(xs), y0f = floorf(ys);
  int   x0 = (int)x0f,    y0 = (int)y0f;
  float wx1 = xs - x0f, wx0 = 1.0f - wx1;
  float wy1 = ys - y0f, wy0 = 1.0f - wy1;

  float wt[4]; int off[4]; bool inr[4];
#pragma unroll
  for (int q = 0; q < 4; ++q) {
    int xx = x0 + (q & 1);
    int yy = y0 + (q >> 1);
    bool valid = (xx >= 0) && (xx < kMD) && (yy >= 0) && (yy < kMD);
    bool reg   = valid && (yy >= 120) && (yy < 220) && (xx >= 70) && (xx < 170);
    wt[q]  = (valid ? 1.0f : 0.0f) * ((q & 1) ? wx1 : wx0) * ((q >> 1) ? wy1 : wy0);
    inr[q] = reg;
    off[q] = reg ? ((yy - 120) * 100 + (xx - 70)) : 0;
  }
  for (int ch = 0; ch < kC; ++ch) {
    const float* pp = proj + (size_t)ch * 10000;
    float acc = 0.0f;
#pragma unroll
    for (int q = 0; q < 4; ++q) acc += wt[q] * (inr[q] ? pp[off[q]] : 0.0f);
    rotated[(size_t)ch * (kMD * kMD) + t] = acc;
  }
}

// ---------------------------------------------------------------- translate + max
__global__ void k_trans(const float* __restrict__ rotated, const float* __restrict__ params,
                        const float* __restrict__ maps_last, float* __restrict__ out, int b) {
  int t = blockIdx.x * blockDim.x + threadIdx.x;
  if (t >= kMD * kMD) return;
  int i = t / kMD, j = t % kMD;
  float dx = params[b * 4 + 2], dy = params[b * 4 + 3];
  float fx = floorf(dx), fy = floorf(dy);
  int   x0 = j + (int)fx, y0 = i + (int)fy;
  float wx1 = dx - fx, wx0 = 1.0f - wx1;
  float wy1 = dy - fy, wy0 = 1.0f - wy1;

  float wt[4]; int off[4]; bool vld[4];
#pragma unroll
  for (int q = 0; q < 4; ++q) {
    int xx = x0 + (q & 1);
    int yy = y0 + (q >> 1);
    bool valid = (xx >= 0) && (xx < kMD) && (yy >= 0) && (yy < kMD);
    wt[q]  = ((q & 1) ? wx1 : wx0) * ((q >> 1) ? wy1 : wy0);
    vld[q] = valid;
    off[q] = valid ? (yy * kMD + xx) : 0;
  }
  for (int ch = 0; ch < kC; ++ch) {
    const float* rp = rotated + (size_t)ch * (kMD * kMD);
    float acc = 0.0f;
#pragma unroll
    for (int q = 0; q < 4; ++q) acc += wt[q] * (vld[q] ? rp[off[q]] : 0.0f);
    size_t oi = (((size_t)b * kC + ch) * kMD + i) * kMD + j;
    out[OUT_MAP + oi] = fmaxf(maps_last[oi], acc);
  }
}

// ---------------------------------------------------------------- launcher
extern "C" void kernel_launch(void* const* d_in, const int* in_sizes, int n_in,
                              void* d_out, int out_size, void* d_ws, size_t ws_size,
                              hipStream_t stream) {
  const float* obs        = (const float*)d_in[0];  // (8,20,120,160)
  const float* pose_obs   = (const float*)d_in[1];  // (8,3)
  const float* maps_last  = (const float*)d_in[2];  // (8,20,240,240)
  const float* poses_last = (const float*)d_in[3];  // (8,3)
  float* out = (float*)d_out;
  float* ws  = (float*)d_ws;

  float* vox    = ws + WS_VOX;
  float* proj   = ws + WS_PROJ;
  float* rot    = ws + WS_ROT;
  float* params = ws + WS_PAR;

  k_pose<<<1, 32, 0, stream>>>(pose_obs, poses_last, out, params);

  const int zero_n4 = (int)((VOX_N + PROJ_N) / 4);             // 3,450,000
  const int zero_bl = (zero_n4 + 255) / 256;
  const int splat_bl = (kP * kF) / 256;                        // 1275
  const int proj_bl  = kF * 625;                               // 10625 waves
  const int pix_bl   = (kMD * kMD) / 256;                      // 225

  for (int b = 0; b < kB; ++b) {
    k_zero <<<zero_bl, 256, 0, stream>>>((float4*)ws, zero_n4);
    k_splat<<<splat_bl, 256, 0, stream>>>(obs, vox, b);
    k_proj <<<proj_bl, 32, 0, stream>>>(vox, proj, out, b);
    k_rot  <<<pix_bl, 256, 0, stream>>>(proj, params, rot, b);
    k_trans<<<pix_bl, 256, 0, stream>>>(rot, params, maps_last, out, b);
  }
}
